// GraphAttentionLayer_2061584302120
// MI455X (gfx1250) — matrix operand order repaired and checked by reading, not compiled, not run on hardware
//
#include <hip/hip_runtime.h>

#define B_   16
#define N_   2048
#define FIN  256
#define FOUT 64
#define NEGV (-9000000000000000.0f)

typedef __attribute__((ext_vector_type(16))) __bf16       v16bf;
typedef __attribute__((ext_vector_type(8)))  float        v8f;
typedef __attribute__((ext_vector_type(8)))  unsigned int v8u;

static __device__ __forceinline__ unsigned short bf16bits(float f) {
  return __builtin_bit_cast(unsigned short, (__bf16)f);
}
template <int IMM>
static __device__ __forceinline__ float swz(float x) {
  return __builtin_bit_cast(float,
      __builtin_amdgcn_ds_swizzle(__builtin_bit_cast(int, x), IMM));
}
static __device__ __forceinline__ float swz_xor16(float x) {
  return swz<(16 << 10) | 0x1f>(x);   // lane ^ 16
}
static __device__ __forceinline__ float red16(float x) {
  x += swz<(1 << 10) | 0x1f>(x);
  x += swz<(2 << 10) | 0x1f>(x);
  x += swz<(4 << 10) | 0x1f>(x);
  x += swz<(8 << 10) | 0x1f>(x);
  return x;
}
static __device__ __forceinline__ float bperm(float x, int srclane) {
  return __builtin_bit_cast(float,
      __builtin_amdgcn_ds_bpermute(srclane << 2, __builtin_bit_cast(int, x)));
}
// CDNA5 async copy: global -> LDS, per-lane LDS dest address, GVS addressing.
static __device__ __forceinline__ void async_ld_b128(unsigned ldsaddr, unsigned voff,
                                                     const void* saddr) {
  asm volatile("global_load_async_to_lds_b128 %0, %1, %2"
               :: "v"(ldsaddr), "v"(voff), "s"(saddr) : "memory");
}
static __device__ __forceinline__ void wait_async0() {
  asm volatile("s_wait_asynccnt 0x0" ::: "memory");
}

// ---------------- kernel 0a: x (fp32) -> xbf (bf16 row-major) ----------------
__global__ __launch_bounds__(256) void k_cvt_x(const float* __restrict__ x,
                                               unsigned int* __restrict__ xbf2) {
  long tid = (long)blockIdx.x * 256 + threadIdx.x;   // 4 floats each
  const float4 v = ((const float4*)x)[tid];
  unsigned int u0 = (unsigned int)bf16bits(v.x) | ((unsigned int)bf16bits(v.y) << 16);
  unsigned int u1 = (unsigned int)bf16bits(v.z) | ((unsigned int)bf16bits(v.w) << 16);
  ((uint2*)xbf2)[tid] = make_uint2(u0, u1);
}

// ------- kernel 0b: weight (256x64 fp32) -> WMMA-B blocked bf16 layout -------
__global__ void k_build_w(const float* __restrict__ w, unsigned int* __restrict__ wblk) {
  int kt = blockIdx.x >> 2, ot = blockIdx.x & 3;
  int lane = threadIdx.x;
  int c = ot * 16 + (lane & 15);
  int koff = (lane & 16) ? 16 : 0;
#pragma unroll
  for (int v = 0; v < 8; ++v) {
    int k0 = kt * 32 + 2 * v + koff;
    unsigned int lo = bf16bits(w[k0 * FOUT + c]);
    unsigned int hi = bf16bits(w[(k0 + 1) * FOUT + c]);
    wblk[((size_t)blockIdx.x * 32 + lane) * 8 + v] = lo | (hi << 16);
  }
}

// ---- kernel 1: h = x@W via WMMA bf16; emit hbf (bf16) + s_src/s_dst (f32) ----
__global__ __launch_bounds__(256) void k_gemm(const unsigned int* __restrict__ xbf2,
                                              const unsigned int* __restrict__ wblk,
                                              const float* __restrict__ w2,
                                              unsigned short* __restrict__ hbf,
                                              float* __restrict__ ssrc,
                                              float* __restrict__ sdst) {
  const int lane = threadIdx.x & 31;
  const int wid  = threadIdx.x >> 5;
  const long tile = (long)blockIdx.x * 8 + wid;
  const long r0 = tile * 16;
  const int hi = (lane >> 4) & 1;
  const int n  = lane & 15;

  v8f z = {0.f, 0.f, 0.f, 0.f, 0.f, 0.f, 0.f, 0.f};
  v8f acc[4] = {z, z, z, z};

  const long rowbase = (r0 + n) * (FIN / 2);
#pragma unroll
  for (int kt = 0; kt < 8; ++kt) {
    v8u au;
#pragma unroll
    for (int v = 0; v < 8; ++v) {
      int kd = kt * 16 + v + hi * 8;
      au[v] = xbf2[rowbase + kd];
    }
    v16bf A = __builtin_bit_cast(v16bf, au);
#pragma unroll
    for (int ot = 0; ot < 4; ++ot) {
      v8u bu = *(const v8u*)(wblk + (((size_t)(kt * 4 + ot) * 32 + lane) * 8));
      v16bf Bt = __builtin_bit_cast(v16bf, bu);
      acc[ot] = __builtin_amdgcn_wmma_f32_16x16x32_bf16(
          false, A, false, Bt, (short)0, acc[ot], false, false);
    }
  }

#pragma unroll
  for (int ot = 0; ot < 4; ++ot)
#pragma unroll
    for (int g = 0; g < 8; ++g) {
      int row = g + hi * 8;
      hbf[(r0 + row) * FOUT + ot * 16 + n] = bf16bits(acc[ot][g]);
    }

  float as[4], ad[4];
#pragma unroll
  for (int ot = 0; ot < 4; ++ot) {
    as[ot] = w2[ot * 16 + n];
    ad[ot] = w2[FOUT + ot * 16 + n];
  }
#pragma unroll
  for (int g = 0; g < 8; ++g) {
    float ps = acc[0][g] * as[0] + acc[1][g] * as[1] + acc[2][g] * as[2] + acc[3][g] * as[3];
    float pd = acc[0][g] * ad[0] + acc[1][g] * ad[1] + acc[2][g] * ad[2] + acc[3][g] * ad[3];
    ps = red16(ps);
    pd = red16(pd);
    if (n == 0) {
      int row = g + hi * 8;
      ssrc[r0 + row] = ps;
      sdst[r0 + row] = pd;
    }
  }
}

// ------ kernel 2: fused masked softmax + (attn @ h), async double-buffered ------
__global__ __launch_bounds__(256) void k_attn(const int* __restrict__ adj,
                                              const unsigned short* __restrict__ hbf,
                                              const float* __restrict__ ssrc,
                                              const float* __restrict__ sdst,
                                              float* __restrict__ out) {
  __shared__ __align__(16) unsigned int  sh_adj[2][8 * 512];   // 2 x 16 KB
  __shared__ unsigned short sh_hT[2][FOUT * 32];               // 2 x  4 KB
  __shared__ float sh_sdt[2][32];

  const int t = threadIdx.x;
  const int lane = t & 31, wid = t >> 5;
  const int tile = blockIdx.x * 8 + wid;
  const int b  = tile >> 7;
  const int i0 = (tile & 127) << 4;
  const int hi = (lane >> 4) & 1;
  const int m  = lane & 15;
  const long bN = (long)b * N_;

  const float ss = ssrc[bN + i0 + m];

  float runM = -__builtin_inff(), runS = 0.f;
  v8f z = {0.f, 0.f, 0.f, 0.f, 0.f, 0.f, 0.f, 0.f};
  v8f acc[4] = {z, z, z, z};

  int ke[16];
#pragma unroll
  for (int e = 0; e < 16; ++e) ke[e] = e + hi * 16;

  const int sk  = t >> 3;
  const int sc0 = (t & 7) << 3;

  // per-lane constant async-copy addresses (LDS dest + global byte offset)
  const unsigned lds0 = (unsigned)(uintptr_t)(&sh_adj[0][0]);
  unsigned adjb[4], voff[4];
#pragma unroll
  for (int q = 0; q < 4; ++q) {
    int lin = q * 128 + lane * 4;
    int row = lin >> 5, col = lin & 31;
    adjb[q] = lds0 + (unsigned)(wid * 512 + row * 32 + (col ^ ((row & 7) << 2))) * 4u;
    voff[q] = (unsigned)(row * N_ + col) * 4u;
  }
  const int* abase0 = adj + (bN + i0) * (long)N_;

  // ---- prologue: stage tile 0 ----
  {
    const uint4 v = *(const uint4*)(hbf + ((bN + sk) * FOUT + sc0));
    unsigned wd[4] = {v.x, v.y, v.z, v.w};
#pragma unroll
    for (int i = 0; i < 8; ++i)
      sh_hT[0][(sc0 + i) * 32 + sk] = (unsigned short)(wd[i >> 1] >> (16 * (i & 1)));
    if (t < 32) sh_sdt[0][t] = sdst[bN + t];
#pragma unroll
    for (int q = 0; q < 4; ++q) async_ld_b128(adjb[q], voff[q], abase0);
    wait_async0();
  }
  __syncthreads();

  const int NT = N_ / 32;
  for (int jt = 0; jt < NT; ++jt) {
    const int buf = jt & 1, nxt = buf ^ 1;
    const bool more = (jt + 1) < NT;
    const long j0n = (long)(jt + 1) * 32;

    // ---- issue next tile's loads early (hide behind compute) ----
    uint4 hv;
    if (more) {
      hv = *(const uint4*)(hbf + ((bN + j0n + sk) * FOUT + sc0));
      const int* sa = abase0 + j0n;
#pragma unroll
      for (int q = 0; q < 4; ++q)
        async_ld_b128(adjb[q] + (unsigned)nxt * (8u * 512u * 4u), voff[q], sa);
    }

    const unsigned* adjw = &sh_adj[buf][wid * 512];
    const unsigned* hTu  = (const unsigned*)sh_hT[buf];
    const float*    sdt  = sh_sdt[buf];

    // ---- masked scores in WMMA-A layout + online softmax ----
    float s[16];
    float tmax = -__builtin_inff();
#pragma unroll
    for (int e = 0; e < 16; ++e) {
      int k = ke[e];
      int a = (int)adjw[m * 32 + (k ^ ((m & 7) << 2))];
      float sc = (a > 0) ? fmaxf(ss + sdt[k], 0.f) : NEGV;
      s[e] = sc;
      tmax = fmaxf(tmax, sc);
    }
    tmax = fmaxf(tmax, swz_xor16(tmax));
    float newM = fmaxf(runM, tmax);
    float corr = __expf(runM - newM);
    float ls = 0.f;
    v16bf pa;
#pragma unroll
    for (int e = 0; e < 16; ++e) {
      float p = __expf(s[e] - newM);
      ls += p;
      pa[e] = (__bf16)p;
    }
    ls += swz_xor16(ls);
    runS = runS * corr + ls;
    runM = newM;

#pragma unroll
    for (int g = 0; g < 8; ++g) {
      float cg = bperm(corr, g + hi * 8);
      acc[0][g] *= cg; acc[1][g] *= cg; acc[2][g] *= cg; acc[3][g] *= cg;
    }

#pragma unroll
    for (int ot = 0; ot < 4; ++ot) {
      v8u bu;
#pragma unroll
      for (int v = 0; v < 8; ++v)
        bu[v] = hTu[(ot * 16 + m) * 16 + v + hi * 8];
      v16bf Bt = __builtin_bit_cast(v16bf, bu);
      acc[ot] = __builtin_amdgcn_wmma_f32_16x16x32_bf16(
          false, pa, false, Bt, (short)0, acc[ot], false, false);
    }

    // ---- store next h tile (transposed) into the other buffer ----
    if (more) {
      unsigned wd[4] = {hv.x, hv.y, hv.z, hv.w};
#pragma unroll
      for (int i = 0; i < 8; ++i)
        sh_hT[nxt][(sc0 + i) * 32 + sk] = (unsigned short)(wd[i >> 1] >> (16 * (i & 1)));
      if (t < 32) sh_sdt[nxt][t] = sdst[bN + j0n + t];
    }
    wait_async0();
    __syncthreads();
  }

  // ---- epilogue ----
#pragma unroll
  for (int g = 0; g < 8; ++g) {
    float inv = 1.0f / bperm(runS, g + hi * 8);
    int row = g + hi * 8;
    long obase = (bN + i0 + row) * FOUT + m;
    out[obase +  0] = acc[0][g] * inv;
    out[obase + 16] = acc[1][g] * inv;
    out[obase + 32] = acc[2][g] * inv;
    out[obase + 48] = acc[3][g] * inv;
  }
}

extern "C" void kernel_launch(void* const* d_in, const int* in_sizes, int n_in,
                              void* d_out, int out_size, void* d_ws, size_t ws_size,
                              hipStream_t stream) {
  const float* x   = (const float*)d_in[0];
  const int*   adj = (const int*)d_in[1];
  const float* w   = (const float*)d_in[2];
  const float* w2  = (const float*)d_in[3];
  float* out = (float*)d_out;

  char* ws = (char*)d_ws;
  const size_t OFF_XBF  = 0;
  const size_t OFF_HBF  = 16u * 1024 * 1024;
  const size_t OFF_WBLK = 20u * 1024 * 1024;
  const size_t OFF_SSRC = OFF_WBLK + (64u << 10);
  const size_t OFF_SDST = OFF_SSRC + (128u << 10);

  unsigned int*   xbf2 = (unsigned int*)(ws + OFF_XBF);
  unsigned short* hbf  = (unsigned short*)(ws + OFF_HBF);
  unsigned int*   wblk = (unsigned int*)(ws + OFF_WBLK);
  float*          ssrc = (float*)(ws + OFF_SSRC);
  float*          sdst = (float*)(ws + OFF_SDST);

  k_cvt_x<<<(B_ * N_ * FIN) / 4 / 256, 256, 0, stream>>>(x, xbf2);
  k_build_w<<<32, 32, 0, stream>>>(w, wblk);
  k_gemm<<<(B_ * N_ / 16) / 8, 256, 0, stream>>>(xbf2, wblk, w2, hbf, ssrc, sdst);
  k_attn<<<(B_ * (N_ / 16)) / 8, 256, 0, stream>>>(adj, hbf, ssrc, sdst, out);
}
